// CommonCrossAttentionWeights_28578712387899
// MI455X (gfx1250) — matrix operand order repaired and checked by reading, not compiled, not run on hardware
//
#include <hip/hip_runtime.h>

// ---------------------------------------------------------------------------
// CDNA5 (gfx1250) bf16 WMMA implementation of CommonCrossAttentionWeights.
// B=2, N_AMR=512, N_VIS=4096, D=1024, H=16, HEAD=64, SCALE=0.125
// ---------------------------------------------------------------------------

typedef __attribute__((ext_vector_type(16))) __bf16 v16bf;
typedef __attribute__((ext_vector_type(8)))  float  v8f;

union FragU { uint4 q[2]; v16bf v; };

__device__ __forceinline__ __bf16 to_bf16(float f) {
    unsigned u = __builtin_bit_cast(unsigned, f);
    u += 0x7FFFu + ((u >> 16) & 1u);               // round-to-nearest-even
    unsigned short h = (unsigned short)(u >> 16);
    return __builtin_bit_cast(__bf16, h);
}

__device__ __forceinline__ v8f wmma_bf16(v16bf a, v16bf b, v8f c) {
    // D = A(16x32 bf16) * B(32x16 bf16) + C(16x16 f32)
    return __builtin_amdgcn_wmma_f32_16x16x32_bf16(
        /*neg_a=*/false, a, /*neg_b=*/false, b,
        /*c_mod=*/(short)0, c, /*reuse_a=*/false, /*reuse_b=*/false);
}

// CDNA5 async global->LDS copy (16B per lane), tracked with ASYNCcnt.
// lds_addr = low 32 bits of flat shared-pointer (LDS aperture => addr[31:0]).
__device__ __forceinline__ void async_ld_b128(unsigned lds_addr, const void* gptr) {
    asm volatile("global_load_async_to_lds_b128 %0, %1, off"
                 :: "v"(lds_addr), "v"(gptr) : "memory");
}
__device__ __forceinline__ void wait_async0() {
    asm volatile("s_wait_asynccnt 0x0" ::: "memory");
}
__device__ __forceinline__ unsigned lds_off(const void* p) {
    return (unsigned)(size_t)p;
}

// A fragment: 16 rows x 32 K, LDS row-major [row][k], stride in bf16 units
// lane<16 : row = row0+lane, K {0..7,16..23};  lane>=16: row0+lane-16, K {8..15,24..31}
__device__ __forceinline__ v16bf frag_a(const __bf16* lds, int stride, int row0, int k0) {
    const int lane = threadIdx.x & 31;
    const int hlf  = lane >> 4;
    const __bf16* base = lds + (size_t)(row0 + (lane & 15)) * stride + k0;
    FragU u;
    u.q[0] = *reinterpret_cast<const uint4*>(base + hlf * 8);
    u.q[1] = *reinterpret_cast<const uint4*>(base + 16 + hlf * 8);
    return u.v;
}

// B fragment: 32 K x 16 cols, LDS stored [col][k] (K contiguous per column)
// lane<16 : col = col0+lane, K {0..7,16..23};  lane>=16: col0+lane-16, K {8..15,24..31}
__device__ __forceinline__ v16bf frag_b(const __bf16* lds, int stride, int col0, int k0) {
    const int lane = threadIdx.x & 31;
    const int hlf  = lane >> 4;
    const __bf16* base = lds + (size_t)(col0 + (lane & 15)) * stride + k0;
    FragU u;
    u.q[0] = *reinterpret_cast<const uint4*>(base + hlf * 8);
    u.q[1] = *reinterpret_cast<const uint4*>(base + 16 + hlf * 8);
    return u.v;
}

// ---------------------------------------------------------------------------
// f32 -> bf16 conversion, 8 elements / thread (float4 x2 in, uint4 out)
// ---------------------------------------------------------------------------
__global__ void cvt_bf16_kernel(const float* __restrict__ in,
                                __bf16* __restrict__ out, int n8) {
    int i = blockIdx.x * blockDim.x + threadIdx.x;
    if (i >= n8) return;
    const float4* p = reinterpret_cast<const float4*>(in) + (size_t)i * 2;
    float4 x0 = p[0], x1 = p[1];
    union { uint4 q; __bf16 h[8]; } u;
    u.h[0] = to_bf16(x0.x); u.h[1] = to_bf16(x0.y);
    u.h[2] = to_bf16(x0.z); u.h[3] = to_bf16(x0.w);
    u.h[4] = to_bf16(x1.x); u.h[5] = to_bf16(x1.y);
    u.h[6] = to_bf16(x1.z); u.h[7] = to_bf16(x1.w);
    reinterpret_cast<uint4*>(out)[i] = u.q;
}

// ---------------------------------------------------------------------------
// Tiled bf16 GEMM: C[M,N] = A[M,K] @ B[K,N]
// Block tile 128x128, K-step 32, double-buffered LDS software pipeline:
// async A-copy + B-transpose for tile k+1 issued under the WMMAs of tile k,
// one s_wait_asynccnt + one barrier per iteration.
// 8 waves: 4(m) x 2(n), each wave 32x64.
// OUT_F32: f32 out with fused +bias[N] and *gate epilogue; else bf16 out.
// ---------------------------------------------------------------------------
#define GLDS 40   // LDS stride (bf16 units); 40*2=80B, multiple of 16B

template<bool OUT_F32>
__global__ __launch_bounds__(256) void gemm_bf16_kernel(
    const __bf16* __restrict__ A, const __bf16* __restrict__ Bw,
    __bf16* __restrict__ Cb, float* __restrict__ Cf,
    const float* __restrict__ bias, const float* __restrict__ gate,
    int M, int N, int K)
{
    __shared__ __bf16 sA[2][128 * GLDS];
    __shared__ __bf16 sB[2][128 * GLDS];   // transposed: [n][k]

    const int tid  = threadIdx.x;
    const int lane = tid & 31;
    const int w    = tid >> 5;
    const int wm   = (w & 3) * 32;      // wave row base within tile
    const int wn   = (w >> 2) * 64;     // wave col base within tile
    const int hlf  = lane >> 4;
    const int cl   = lane & 15;
    const int m0   = blockIdx.y * 128;
    const int n0   = blockIdx.x * 128;

    v8f zero = {0.f,0.f,0.f,0.f,0.f,0.f,0.f,0.f};
    v8f acc[2][4];
    #pragma unroll
    for (int i = 0; i < 2; i++)
        #pragma unroll
        for (int j = 0; j < 4; j++) acc[i][j] = zero;

    // per-thread staging coordinates
    const int rA = tid >> 1;
    const int cA = (tid & 1) * 16;
    const unsigned ldsA[2] = { lds_off(&sA[0][rA * GLDS + cA]),
                               lds_off(&sA[1][rA * GLDS + cA]) };
    const int kB = tid >> 3;            // 0..31
    const int nB = (tid & 7) * 16;      // 0..112
    const __bf16* gA = A + (size_t)(m0 + rA) * K + cA;

    auto stage = [&](int buf, int k0) {
        // A tile 128x32 via CDNA5 async global->LDS (2 x b128 / lane)
        const __bf16* g = gA + k0;
        async_ld_b128(ldsA[buf],      g);
        async_ld_b128(ldsA[buf] + 16, g + 8);
        if (k0 + 32 < K) __builtin_prefetch(g + 64, 0, 3);   // k+2 tile
        // B tile 32x128 transposed into sB[buf][n][k]
        const __bf16* gb = Bw + (size_t)(k0 + kB) * N + n0 + nB;
        FragU u;
        u.q[0] = *reinterpret_cast<const uint4*>(gb);
        u.q[1] = *reinterpret_cast<const uint4*>(gb + 8);
        __bf16* sBb = sB[buf];
        #pragma unroll
        for (int j = 0; j < 16; j++) sBb[(nB + j) * GLDS + kB] = u.v[j];
    };

    // prologue
    stage(0, 0);
    wait_async0();
    __syncthreads();

    const int kIters = K / 32;
    for (int kt = 0; kt < kIters; kt++) {
        const int cur = kt & 1;
        if (kt + 1 < kIters) stage(cur ^ 1, (kt + 1) * 32);

        const __bf16* cA_ = sA[cur];
        const __bf16* cB_ = sB[cur];
        v16bf a0 = frag_a(cA_, GLDS, wm,      0);
        v16bf a1 = frag_a(cA_, GLDS, wm + 16, 0);
        #pragma unroll
        for (int ni = 0; ni < 4; ni++) {
            v16bf bf = frag_b(cB_, GLDS, wn + ni * 16, 0);
            acc[0][ni] = wmma_bf16(a0, bf, acc[0][ni]);
            acc[1][ni] = wmma_bf16(a1, bf, acc[1][ni]);
        }
        wait_async0();
        __syncthreads();
    }

    // epilogue: C row = base + v + (hlf?8:0), col = base + lane&15
    #pragma unroll
    for (int mi = 0; mi < 2; mi++)
        #pragma unroll
        for (int ni = 0; ni < 4; ni++)
            #pragma unroll
            for (int v = 0; v < 8; v++) {
                int row = m0 + wm + mi * 16 + v + hlf * 8;
                int col = n0 + wn + ni * 16 + cl;
                float val = acc[mi][ni][v];
                size_t idx = (size_t)row * N + col;
                if constexpr (OUT_F32) {
                    val += bias[col];
                    val *= gate[idx];
                    Cf[idx] = val;
                } else {
                    Cb[idx] = to_bf16(val);
                }
            }
}

// ---------------------------------------------------------------------------
// Streaming cross-attention (two-pass softmax), one (b, h, 64-row block) per
// workgroup. Q/K/V are bf16 [B*rows, 1024] row-major; head h = cols h*64..+63.
// Out = softmax(Q K^T * scale  [masked over cols]) @ V, written bf16.
// HAS_MASK selects the pad-masked direction (branch-free via LDS keep[]).
// ---------------------------------------------------------------------------
template<bool HAS_MASK>
__global__ __launch_bounds__(256) void attn_kernel(
    const __bf16* __restrict__ Qg, const __bf16* __restrict__ Kg,
    const __bf16* __restrict__ Vg, __bf16* __restrict__ Og,
    const unsigned char* __restrict__ mask,   // [B, ncols] (HAS_MASK only)
    int nrows, int ncols, float scale)
{
    const int DD = 1024;
    __shared__ __bf16 sQ [64 * 72];
    __shared__ __bf16 sK [64 * 72];
    __shared__ __bf16 sVt[64 * 72];   // [c][m]
    __shared__ float  sS [64 * 68];
    __shared__ __bf16 sP [64 * 72];
    __shared__ float  sMax[64];
    __shared__ float  sSum[64];
    __shared__ float  sKeep[64];      // 1.0 = attendable, 0.0 = padded out

    const int tid  = threadIdx.x;
    const int lane = tid & 31;
    const int w    = tid >> 5;
    const int wm   = (w & 3) * 16;    // S/O row base
    const int wn   = (w >> 2) * 32;   // S/O col base
    const int hlf  = lane >> 4;
    const int cl   = lane & 15;

    const int m0 = blockIdx.x * 64;
    const int h  = blockIdx.y;
    const int b  = blockIdx.z;
    const size_t qbase = (size_t)b * nrows * DD + h * 64;
    const size_t kbase = (size_t)b * ncols * DD + h * 64;
    const float sl2e = scale * 1.44269504f;

    const int rT = tid >> 2;            // 0..63
    const int cT = (tid & 3) * 16;      // 0..48
    const unsigned ldsQ = lds_off(&sQ[rT * 72 + cT]);
    const unsigned ldsK = lds_off(&sK[rT * 72 + cT]);

    // Q block -> LDS via async copy (resident for whole kernel)
    {
        const __bf16* g = Qg + qbase + (size_t)(m0 + rT) * DD + cT;
        async_ld_b128(ldsQ,      g);
        async_ld_b128(ldsQ + 16, g + 8);
    }
    if (tid < 64) { sMax[tid] = -3.0e38f; sSum[tid] = 0.0f; }
    wait_async0();
    __syncthreads();

    const int ntiles = ncols / 64;
    v8f zero = {0.f,0.f,0.f,0.f,0.f,0.f,0.f,0.f};

    // ---------------- pass 1: row maxima of raw scores ----------------
    for (int ct = 0; ct < ntiles; ct++) {
        {
            const __bf16* g = Kg + kbase + (size_t)(ct * 64 + rT) * DD + cT;
            async_ld_b128(ldsK,      g);
            async_ld_b128(ldsK + 16, g + 8);
        }
        if (HAS_MASK && tid < 64)
            sKeep[tid] = mask[(size_t)b * ncols + ct * 64 + tid] ? 0.0f : 1.0f;
        wait_async0();
        __syncthreads();

        v8f s0 = zero, s1 = zero;
        #pragma unroll
        for (int k = 0; k < 64; k += 32) {
            v16bf a  = frag_a(sQ, 72, wm, k);
            v16bf b0 = frag_b(sK, 72, wn,      k);
            v16bf b1 = frag_b(sK, 72, wn + 16, k);
            s0 = wmma_bf16(a, b0, s0);
            s1 = wmma_bf16(a, b1, s1);
        }
        #pragma unroll
        for (int v = 0; v < 8; v++) {
            int rl = wm + v + hlf * 8;
            sS[rl * 68 + wn + cl]      = s0[v];
            sS[rl * 68 + wn + 16 + cl] = s1[v];
        }
        __syncthreads();

        if (tid < 64) {
            float mx = sMax[tid];
            for (int j = 0; j < 64; j++) {
                float sv = sS[tid * 68 + j];
                if (HAS_MASK) sv = (sKeep[j] != 0.0f) ? sv : -3.0e38f;
                mx = fmaxf(mx, sv);
            }
            sMax[tid] = mx;
        }
        __syncthreads();
    }

    // ---------------- pass 2: exp-sum + P @ V accumulation ----------------
    v8f o0 = zero, o1 = zero;
    for (int ct = 0; ct < ntiles; ct++) {
        {
            const __bf16* g = Kg + kbase + (size_t)(ct * 64 + rT) * DD + cT;
            async_ld_b128(ldsK,      g);
            async_ld_b128(ldsK + 16, g + 8);
        }
        {   // V tile transposed into sVt[c][m] (VALU path: transpose-on-store)
            int mloc = tid >> 2;
            int c0   = (tid & 3) * 16;
            const __bf16* g = Vg + kbase + (size_t)(ct * 64 + mloc) * DD + c0;
            FragU u;
            u.q[0] = *reinterpret_cast<const uint4*>(g);
            u.q[1] = *reinterpret_cast<const uint4*>(g + 8);
            #pragma unroll
            for (int j = 0; j < 16; j++) sVt[(c0 + j) * 72 + mloc] = u.v[j];
        }
        if (HAS_MASK && tid < 64)
            sKeep[tid] = mask[(size_t)b * ncols + ct * 64 + tid] ? 0.0f : 1.0f;
        wait_async0();
        __syncthreads();

        v8f s0 = zero, s1 = zero;
        #pragma unroll
        for (int k = 0; k < 64; k += 32) {
            v16bf a  = frag_a(sQ, 72, wm, k);
            v16bf b0 = frag_b(sK, 72, wn,      k);
            v16bf b1 = frag_b(sK, 72, wn + 16, k);
            s0 = wmma_bf16(a, b0, s0);
            s1 = wmma_bf16(a, b1, s1);
        }
        #pragma unroll
        for (int v = 0; v < 8; v++) {
            int rl = wm + v + hlf * 8;
            sS[rl * 68 + wn + cl]      = s0[v];
            sS[rl * 68 + wn + 16 + cl] = s1[v];
        }
        __syncthreads();

        if (tid < 64) {
            float mx  = sMax[tid];
            float sum = sSum[tid];
            for (int j = 0; j < 64; j++) {
                float p = exp2f((sS[tid * 68 + j] - mx) * sl2e);
                if (HAS_MASK) p = (sKeep[j] != 0.0f) ? p : 0.0f;
                sum += p;
                sP[tid * 72 + j] = to_bf16(p);
            }
            sSum[tid] = sum;
        }
        __syncthreads();

        #pragma unroll
        for (int kk = 0; kk < 64; kk += 32) {
            v16bf a  = frag_a(sP, 72, wm, kk);
            v16bf b0 = frag_b(sVt, 72, wn,      kk);
            v16bf b1 = frag_b(sVt, 72, wn + 16, kk);
            o0 = wmma_bf16(a, b0, o0);
            o1 = wmma_bf16(a, b1, o1);
        }
        __syncthreads();
    }

    // normalize and write O (bf16, layout [b, row, h*64 + c])
    #pragma unroll
    for (int v = 0; v < 8; v++) {
        int rl = wm + v + hlf * 8;
        float inv = 1.0f / sSum[rl];
        size_t rowoff = qbase + (size_t)(m0 + rl) * DD;
        Og[rowoff + wn + cl]      = to_bf16(o0[v] * inv);
        Og[rowoff + wn + 16 + cl] = to_bf16(o1[v] * inv);
    }
}

// ---------------------------------------------------------------------------
// Host launch
// ---------------------------------------------------------------------------
extern "C" void kernel_launch(void* const* d_in, const int* in_sizes, int n_in,
                              void* d_out, int out_size, void* d_ws, size_t ws_size,
                              hipStream_t stream) {
    (void)in_sizes; (void)n_in; (void)out_size; (void)ws_size;

    const int B = 2, N_AMR = 512, N_VIS = 4096, D = 1024, H = 16;
    const float SCALE = 0.125f;   // 64^-0.5
    const size_t nAmrD = (size_t)B * N_AMR * D;   // 1,048,576
    const size_t nVisD = (size_t)B * N_VIS * D;   // 8,388,608
    const size_t nW    = (size_t)D * D;           // 1,048,576

    const float* amr  = (const float*)d_in[0];
    const unsigned char* padmask = (const unsigned char*)d_in[1];
    const float* vis  = (const float*)d_in[2];
    const float* W_amr     = (const float*)d_in[3];
    const float* W_amr_v   = (const float*)d_in[4];
    const float* W_vis     = (const float*)d_in[5];
    const float* W_vis_v   = (const float*)d_in[6];
    const float* W_amr_out = (const float*)d_in[7];
    const float* b_amr_out = (const float*)d_in[8];
    const float* W_vis_out = (const float*)d_in[9];
    const float* b_vis_out = (const float*)d_in[10];

    char* ws = (char*)d_ws;
    size_t off = 0;
    auto alloc = [&](size_t bytes) -> void* {
        void* p = ws + off;
        off = (off + bytes + 255) & ~(size_t)255;
        return p;
    };

    __bf16* bAmr     = (__bf16*)alloc(nAmrD * 2);
    __bf16* bVis     = (__bf16*)alloc(nVisD * 2);
    __bf16* bWamr    = (__bf16*)alloc(nW * 2);
    __bf16* bWamrV   = (__bf16*)alloc(nW * 2);
    __bf16* bWvis    = (__bf16*)alloc(nW * 2);
    __bf16* bWvisV   = (__bf16*)alloc(nW * 2);
    __bf16* bWamrOut = (__bf16*)alloc(nW * 2);
    __bf16* bWvisOut = (__bf16*)alloc(nW * 2);
    __bf16* aq       = (__bf16*)alloc(nAmrD * 2);
    __bf16* av       = (__bf16*)alloc(nAmrD * 2);
    __bf16* vq       = (__bf16*)alloc(nVisD * 2);
    __bf16* vv       = (__bf16*)alloc(nVisD * 2);
    __bf16* amr2pre  = (__bf16*)alloc(nAmrD * 2);
    __bf16* vis2pre  = (__bf16*)alloc(nVisD * 2);

    // 1) convert inputs/weights to bf16
    auto cvt = [&](const float* src, __bf16* dst, size_t n) {
        int n8 = (int)(n / 8);
        cvt_bf16_kernel<<<(n8 + 255) / 256, 256, 0, stream>>>(src, dst, n8);
    };
    cvt(amr, bAmr, nAmrD);
    cvt(vis, bVis, nVisD);
    cvt(W_amr, bWamr, nW);
    cvt(W_amr_v, bWamrV, nW);
    cvt(W_vis, bWvis, nW);
    cvt(W_vis_v, bWvisV, nW);
    cvt(W_amr_out, bWamrOut, nW);
    cvt(W_vis_out, bWvisOut, nW);

    // 2) projections (bf16 out)
    dim3 blk(256);
    dim3 gAmr(D / 128, (B * N_AMR) / 128);   // (8, 8)
    dim3 gVis(D / 128, (B * N_VIS) / 128);   // (8, 64)
    gemm_bf16_kernel<false><<<gAmr, blk, 0, stream>>>(bAmr, bWamr,  aq, nullptr, nullptr, nullptr, B*N_AMR, D, D);
    gemm_bf16_kernel<false><<<gAmr, blk, 0, stream>>>(bAmr, bWamrV, av, nullptr, nullptr, nullptr, B*N_AMR, D, D);
    gemm_bf16_kernel<false><<<gVis, blk, 0, stream>>>(bVis, bWvis,  vq, nullptr, nullptr, nullptr, B*N_VIS, D, D);
    gemm_bf16_kernel<false><<<gVis, blk, 0, stream>>>(bVis, bWvisV, vv, nullptr, nullptr, nullptr, B*N_VIS, D, D);

    // 3) attention, both directions
    dim3 gA1(N_AMR / 64, H, B);   // amr rows attend over vis cols (no mask)
    attn_kernel<false><<<gA1, blk, 0, stream>>>(aq, vq, vv, amr2pre, nullptr, N_AMR, N_VIS, SCALE);
    dim3 gA2(N_VIS / 64, H, B);   // vis rows attend over amr cols (pad mask)
    attn_kernel<true><<<gA2, blk, 0, stream>>>(vq, aq, av, vis2pre, padmask, N_VIS, N_AMR, SCALE);

    // 4) output projections (+bias, *gate) -> f32 d_out
    float* out_amr = (float*)d_out;
    float* out_vis = out_amr + nAmrD;
    gemm_bf16_kernel<true><<<gAmr, blk, 0, stream>>>(amr2pre, bWamrOut, nullptr, out_amr, b_amr_out, amr, B*N_AMR, D, D);
    gemm_bf16_kernel<true><<<gVis, blk, 0, stream>>>(vis2pre, bWvisOut, nullptr, out_vis, b_vis_out, vis, B*N_VIS, D, D);
}
